// GAT_4011499454637
// MI455X (gfx1250) — matrix operand order repaired and checked by reading, not compiled, not run on hardware
//
#include <hip/hip_runtime.h>
#include <hip/hip_bf16.h>

#define N_NODES 4096
#define NEGBIG  (-9.0e16f)

typedef _Float16 v16h __attribute__((ext_vector_type(16)));
typedef _Float16 v8h  __attribute__((ext_vector_type(8)));
typedef float    v8f  __attribute__((ext_vector_type(8)));

// ---------------------------------------------------------------------------
// Kernel 1: blended autoencoders. x_blend = 0.5*AE1(x) + 0.5*AE2(x)
// ---------------------------------------------------------------------------
__global__ void ae_kernel(const float* __restrict__ x,
                          const float* __restrict__ e1W, const float* __restrict__ e1b,
                          const float* __restrict__ d1W, const float* __restrict__ d1b,
                          const float* __restrict__ e2W, const float* __restrict__ e2b,
                          const float* __restrict__ d2W, const float* __restrict__ d2b,
                          float* __restrict__ outp)
{
    int n = blockIdx.x * blockDim.x + threadIdx.x;
    if (n >= N_NODES) return;
    float xi[5];
#pragma unroll
    for (int k = 0; k < 5; ++k) xi[k] = x[n * 5 + k];

    float o[5];
    float enc[3];
    // AE 1
#pragma unroll
    for (int j = 0; j < 3; ++j) {
        float v = e1b[j];
#pragma unroll
        for (int k = 0; k < 5; ++k) v += xi[k] * e1W[k * 3 + j];
        enc[j] = v > 0.f ? v : 0.05f * v;
    }
#pragma unroll
    for (int j = 0; j < 5; ++j) {
        float v = d1b[j];
#pragma unroll
        for (int k = 0; k < 3; ++k) v += enc[k] * d1W[k * 5 + j];
        o[j] = 0.5f * tanhf(v);
    }
    // AE 2
#pragma unroll
    for (int j = 0; j < 3; ++j) {
        float v = e2b[j];
#pragma unroll
        for (int k = 0; k < 5; ++k) v += xi[k] * e2W[k * 3 + j];
        enc[j] = v > 0.f ? v : 0.05f * v;
    }
#pragma unroll
    for (int j = 0; j < 5; ++j) {
        float v = d2b[j];
#pragma unroll
        for (int k = 0; k < 3; ++k) v += enc[k] * d2W[k * 5 + j];
        o[j] += 0.5f * tanhf(v);
        outp[n * 5 + j] = o[j];
    }
}

// ---------------------------------------------------------------------------
// Kernel 2: htT[h][d][n] = (x @ W1)^T in f16 (transposed so attention B
// fragments are contiguous), plus src/tgt attention scalars.
// ---------------------------------------------------------------------------
__global__ void proj1_kernel(const float* __restrict__ x,     // [N][5]
                             const float* __restrict__ W1,    // [5][1024]
                             const float* __restrict__ a1,    // [4][512]
                             _Float16* __restrict__ htT,      // [4][256][N]
                             float* __restrict__ srcv,        // [4][N]
                             float* __restrict__ tgtv)        // [4][N]
{
    int n = blockIdx.x * blockDim.x + threadIdx.x;
    int h = blockIdx.y;
    if (n >= N_NODES) return;
    float xv[5];
#pragma unroll
    for (int k = 0; k < 5; ++k) xv[k] = x[n * 5 + k];
    const float* aS = a1 + h * 512;
    float s = 0.f, t = 0.f;
    for (int d = 0; d < 256; ++d) {
        float v = 0.f;
#pragma unroll
        for (int k = 0; k < 5; ++k) v += xv[k] * W1[k * 1024 + h * 256 + d];
        htT[((size_t)h * 256 + d) * N_NODES + n] = (_Float16)v;
        s += v * aS[d];
        t += v * aS[256 + d];
    }
    srcv[h * N_NODES + n] = s;
    tgtv[h * N_NODES + n] = t;
}

// ---------------------------------------------------------------------------
// Kernel 3: masked-softmax attention + WMMA aggregation (branchless scans).
// ---------------------------------------------------------------------------
template <int D, int TPW>
__global__ void __launch_bounds__(32 * (D / (16 * TPW)))
attn_kernel(const float* __restrict__ srcv,     // [H][N]
            const float* __restrict__ tgtv,     // [H][N]
            const _Float16* __restrict__ htT,   // [H][D][N] f16 (transposed)
            const int* __restrict__ adj,        // [N][N]
            float* __restrict__ outp)           // [H][N][D]
{
    constexpr int NW   = D / (16 * TPW);
    constexpr int NT   = NW * 32;
    constexpr int NTPR = NT / 16;      // threads per row in softmax pass
    constexpr int PIT  = 512 / NT;     // P-tile elements per thread

    const int it  = blockIdx.x;
    const int h   = blockIdx.y;
    const int i0  = it * 16;
    const int tid = threadIdx.x;
    const int wave = tid >> 5;
    const int lane = tid & 31;

    __shared__ float s_src[16];
    __shared__ float s_m[16], s_linv[16];
    __shared__ float s_pm[16][NTPR];
    __shared__ float s_pl[16][NTPR];
    // P tile, double buffered; rows padded to 40 halves (80B, 16B-aligned,
    // 20-bank rotation -> conflict-free row reads)
    __shared__ __align__(16) _Float16 s_p[2][16][40];

    const float* srcH = srcv + (size_t)h * N_NODES;
    const float* tgtH = tgtv + (size_t)h * N_NODES;
    const _Float16* htTH = htT + (size_t)h * D * N_NODES;

    if (tid < 16) s_src[tid] = srcH[i0 + tid];
    __syncthreads();

    // ---- Pass 1: per-row online softmax statistics over all j (branchless)
    {
        const int row = tid / NTPR;
        const int sub = tid % NTPR;
        const float si = s_src[row];
        const int* adjRow = adj + (size_t)(i0 + row) * N_NODES;
        float m = NEGBIG, l = 0.f;
        for (int j = sub; j < N_NODES; j += NTPR) {
            const int   av = adjRow[j];
            const float tv = tgtH[j];
            float v = si + tv;
            float e = v > 0.f ? v : 0.2f * v;
            e = (av > 0) ? e : NEGBIG;
            if (e > m) { l = l * __expf(m - e) + 1.f; m = e; }
            else       { l += __expf(e - m); }
        }
        s_pm[row][sub] = m;
        s_pl[row][sub] = l;
    }
    __syncthreads();
    if (tid < 16) {
        float m = NEGBIG;
#pragma unroll
        for (int s = 0; s < NTPR; ++s) m = fmaxf(m, s_pm[tid][s]);
        float l = 0.f;
#pragma unroll
        for (int s = 0; s < NTPR; ++s) l += s_pl[tid][s] * __expf(s_pm[tid][s] - m);
        s_m[tid]    = m;
        s_linv[tid] = 1.0f / l;
    }
    __syncthreads();

    // ---- Pass 2: stream j in chunks of 32, WMMA-accumulate P @ ht ----
    v8f acc[TPW];
#pragma unroll
    for (int t = 0; t < TPW; ++t)
#pragma unroll
        for (int q = 0; q < 8; ++q) acc[t][q] = 0.f;

    const int rowA = lane & 15;
    const int kb0  = (lane >> 4) * 16;     // A halves 0..7 -> K kb0..kb0+7
                                           // A halves 8..15 -> K kb0+8..+15
    const int colL = lane & 15;
    const int krL  = (lane >> 4) * 16;     // B halves -> K krL..krL+15

    // per-tile streaming base pointers into transposed ht (column-contiguous)
    const _Float16* bbase[TPW];
#pragma unroll
    for (int t = 0; t < TPW; ++t) {
        const int col = (wave * TPW + t) * 16 + colL;
        bbase[t] = htTH + (size_t)col * N_NODES + krL;
    }

    // fixed (row, jj) assignments for the P-tile fill (compile-time count)
    int pr_[PIT], pj_[PIT];
#pragma unroll
    for (int q = 0; q < PIT; ++q) {
        const int idx = tid + q * NT;
        pr_[q] = idx >> 5;
        pj_[q] = idx & 31;
    }

    for (int j0 = 0; j0 < N_NODES; j0 += 32) {
        const int buf = (j0 >> 5) & 1;
        // P tile: p = exp(e - m) / l, stored f16 (branchless)
#pragma unroll
        for (int q = 0; q < PIT; ++q) {
            const int r  = pr_[q];
            const int jj = pj_[q];
            const int j  = j0 + jj;
            const int   av = adj[(size_t)(i0 + r) * N_NODES + j];
            const float tv = tgtH[j];
            float v = s_src[r] + tv;
            float e = v > 0.f ? v : 0.2f * v;
            e = (av > 0) ? e : NEGBIG;
            const float p = __expf(e - s_m[r]) * s_linv[r];
            s_p[buf][r][jj] = (_Float16)p;
        }
        __syncthreads();   // single barrier per chunk (double-buffered P)

        // A fragment: two aligned 16B LDS loads
        union { v16h v; v8h h[2]; } a;
        {
            const _Float16* pr = &s_p[buf][rowA][0];
            a.h[0] = *(const v8h*)(pr + kb0);
            a.h[1] = *(const v8h*)(pr + kb0 + 8);
        }
#pragma unroll
        for (int t = 0; t < TPW; ++t) {
            union { v16h v; v8h h[2]; } b;
            const _Float16* gp = bbase[t] + j0;
            b.h[0] = *(const v8h*)(gp);
            b.h[1] = *(const v8h*)(gp + 8);
            acc[t] = __builtin_amdgcn_wmma_f32_16x16x32_f16(
                false, a.v, false, b.v, (short)0, acc[t], false, false);
        }
    }

    // ---- Store C tiles: VGPR r -> row (hi*8 + r), lane&15 -> col ----
    float* outH = outp + ((size_t)h * N_NODES + i0) * D;
    const int rbase = (lane >> 4) * 8;
#pragma unroll
    for (int t = 0; t < TPW; ++t) {
        const int n0  = (wave * TPW + t) * 16;
        const int col = n0 + (lane & 15);
#pragma unroll
        for (int r = 0; r < 8; ++r) {
            outH[(size_t)(rbase + r) * D + col] = acc[t][r];
        }
    }
}

// ---------------------------------------------------------------------------
// Kernel 4: x2 = ELU(mean over 4 heads of hp1)
// ---------------------------------------------------------------------------
__global__ void mean_elu_kernel(const float* __restrict__ hp1,  // [4][N][256]
                                float* __restrict__ x2)          // [N][256]
{
    int idx = blockIdx.x * blockDim.x + threadIdx.x;
    if (idx >= N_NODES * 256) return;
    float s = 0.f;
#pragma unroll
    for (int h = 0; h < 4; ++h) s += hp1[(size_t)h * N_NODES * 256 + idx];
    s *= 0.25f;
    x2[idx] = s > 0.f ? s : (__expf(s) - 1.f);
}

// ---------------------------------------------------------------------------
// Kernel 5: ht2T[c][n] = (x2 @ W2)^T (f16) + src2/tgt2. One 64-thread WG/node.
// ---------------------------------------------------------------------------
__global__ void proj2_kernel(const float* __restrict__ x2,   // [N][256]
                             const float* __restrict__ W2,   // [256][64]
                             const float* __restrict__ a2,   // [128]
                             _Float16* __restrict__ ht2T,    // [64][N]
                             float* __restrict__ src2,       // [N]
                             float* __restrict__ tgt2)       // [N]
{
    const int n = blockIdx.x;
    const int c = threadIdx.x;   // 64 threads
    __shared__ float xs[256];
    for (int k = c; k < 256; k += 64) xs[k] = x2[(size_t)n * 256 + k];
    __syncthreads();
    float v = 0.f;
    for (int k = 0; k < 256; ++k) v += xs[k] * W2[k * 64 + c];
    ht2T[(size_t)c * N_NODES + n] = (_Float16)v;

    __shared__ float ps[64], pt[64];
    ps[c] = v * a2[c];
    pt[c] = v * a2[64 + c];
    __syncthreads();
    for (int s = 32; s > 0; s >>= 1) {
        if (c < s) { ps[c] += ps[c + s]; pt[c] += pt[c + s]; }
        __syncthreads();
    }
    if (c == 0) { src2[n] = ps[0]; tgt2[n] = pt[0]; }
}

// ---------------------------------------------------------------------------
// Kernel 6: out = (tanh(hp2)+1)/2
// ---------------------------------------------------------------------------
__global__ void final_kernel(const float* __restrict__ hp2, float* __restrict__ outp)
{
    int idx = blockIdx.x * blockDim.x + threadIdx.x;
    if (idx >= N_NODES * 64) return;
    outp[idx] = (tanhf(hp2[idx]) + 1.f) * 0.5f;
}

// ---------------------------------------------------------------------------
extern "C" void kernel_launch(void* const* d_in, const int* in_sizes, int n_in,
                              void* d_out, int out_size, void* d_ws, size_t ws_size,
                              hipStream_t stream)
{
    (void)in_sizes; (void)n_in; (void)out_size; (void)ws_size;

    const float* x_in = (const float*)d_in[0];
    const int*   adj  = (const int*)d_in[1];
    const float* e1W  = (const float*)d_in[2];
    const float* e1b  = (const float*)d_in[3];
    const float* d1W  = (const float*)d_in[4];
    const float* d1b  = (const float*)d_in[5];
    const float* e2W  = (const float*)d_in[6];
    const float* e2b  = (const float*)d_in[7];
    const float* d2W  = (const float*)d_in[8];
    const float* d2b  = (const float*)d_in[9];
    const float* W1   = (const float*)d_in[10];
    const float* a1   = (const float*)d_in[11];
    const float* W2   = (const float*)d_in[12];
    const float* a2   = (const float*)d_in[13];
    float* outp = (float*)d_out;

    // workspace carve-out (256B aligned)
    char* ws = (char*)d_ws;
    size_t off = 0;
    auto alloc = [&](size_t bytes) -> void* {
        void* p = ws + off;
        off = (off + bytes + 255) & ~(size_t)255;
        return p;
    };
    float*     xb    = (float*)alloc((size_t)N_NODES * 5 * 4);
    float*     src1  = (float*)alloc((size_t)4 * N_NODES * 4);
    float*     tgt1  = (float*)alloc((size_t)4 * N_NODES * 4);
    _Float16*  ht1T  = (_Float16*)alloc((size_t)4 * 256 * N_NODES * 2);
    float*     hp1   = (float*)alloc((size_t)4 * N_NODES * 256 * 4);
    float*     x2    = (float*)alloc((size_t)N_NODES * 256 * 4);
    _Float16*  ht2T  = (_Float16*)alloc((size_t)64 * N_NODES * 2);
    float*     src2  = (float*)alloc((size_t)N_NODES * 4);
    float*     tgt2  = (float*)alloc((size_t)N_NODES * 4);
    float*     hp2   = (float*)alloc((size_t)N_NODES * 64 * 4);

    ae_kernel<<<N_NODES / 256, 256, 0, stream>>>(x_in, e1W, e1b, d1W, d1b,
                                                 e2W, e2b, d2W, d2b, xb);

    proj1_kernel<<<dim3(N_NODES / 256, 4), 256, 0, stream>>>(xb, W1, a1, ht1T, src1, tgt1);

    // GAT layer 1: 4 heads, D=256, 8 waves x 2 tiles each
    attn_kernel<256, 2><<<dim3(N_NODES / 16, 4), 256, 0, stream>>>(src1, tgt1, ht1T, adj, hp1);

    mean_elu_kernel<<<(N_NODES * 256) / 256, 256, 0, stream>>>(hp1, x2);

    proj2_kernel<<<N_NODES, 64, 0, stream>>>(x2, W2, a2, ht2T, src2, tgt2);

    // GAT layer 2: 1 head, D=64, 4 waves x 1 tile each
    attn_kernel<64, 1><<<dim3(N_NODES / 16, 1), 128, 0, stream>>>(src2, tgt2, ht2T, adj, hp2);

    final_kernel<<<(N_NODES * 64) / 256, 256, 0, stream>>>(hp2, outp);
}
